// CryoAttention_15805479649693
// MI455X (gfx1250) — matrix operand order repaired and checked by reading, not compiled, not run on hardware
//
#include <hip/hip_runtime.h>
#include <cstdint>
#include <cstddef>

// ---------------------------------------------------------------------------
// CryoAttention for MI455X (gfx1250): all matmuls via v_wmma_f32_16x16x32_bf16
// A-tiles staged into LDS by the Tensor Data Mover (tensor_load_to_lds),
// double-buffered on TENSORcnt.
// ---------------------------------------------------------------------------

#define DEV __device__ __forceinline__

typedef __attribute__((ext_vector_type(16))) __bf16      bf16x16;
typedef __attribute__((ext_vector_type(8)))  float       floatx8;
typedef __attribute__((ext_vector_type(8)))  unsigned short ushort8;
typedef __attribute__((ext_vector_type(4)))  unsigned int u32x4;
typedef __attribute__((ext_vector_type(8)))  int          i32x8;
typedef __attribute__((ext_vector_type(4)))  int          i32x4;
typedef unsigned short ush;

union Frag16 { ushort8 h[2]; bf16x16 v; };

#define NN    512
#define KK    20
#define NKE   (NN*KK)     // 10240 edges
#define IFZ   256
#define AHZ   8
#define AHIF  2048        // AHZ*IFZ
#define CEQ   1280        // CE*Q
#define P3    4913        // 17^3
#define DD    96
#define EFW   2080        // 2079 padded to mult of 32

DEV ush f2bf(float f) {
  union { float f; unsigned u; } c; c.f = f;
  unsigned u = c.u;
  unsigned r = u + 0x7FFFu + ((u >> 16) & 1u);
  return (ush)(r >> 16);
}

DEV float trilin(const float* __restrict__ g, float pz, float py, float px) {
  float fz = floorf(pz), fy = floorf(py), fx = floorf(px);
  int z0 = (int)fz, y0 = (int)fy, x0 = (int)fx;
  float wz = pz - fz, wy = py - fy, wx = px - fx;
  float out = 0.f;
#pragma unroll
  for (int dz = 0; dz < 2; ++dz) {
    int z = z0 + dz; z = z < 0 ? 0 : (z > DD-1 ? DD-1 : z);
    float az = dz ? wz : 1.f - wz;
#pragma unroll
    for (int dy = 0; dy < 2; ++dy) {
      int y = y0 + dy; y = y < 0 ? 0 : (y > DD-1 ? DD-1 : y);
      float ay = dy ? wy : 1.f - wy;
#pragma unroll
      for (int dx = 0; dx < 2; ++dx) {
        int xx = x0 + dx; xx = xx < 0 ? 0 : (xx > DD-1 ? DD-1 : xx);
        float ax = dx ? wx : 1.f - wx;
        out += az * ay * ax * g[((size_t)z*DD + y)*DD + xx];
      }
    }
  }
  return out;
}

// ---------------------------------------------------------------------------
// TDM: issue a 2D bf16 tile load (tile_w x tile_h elems, row stride
// stride_elems) from global into LDS at lds_byte_off. D# per ISA §8.3/8.4:
// group0 = {count=1 | lds_addr | global_addr[56:0] | type=2},
// group1 = {data_size=2B, tensor_dim0/1 (OOB zero-fill), tile_dim0/1, stride}.
// Tracked on TENSORcnt. 6-arg builtin form (clang-23 / therock headers).
// ---------------------------------------------------------------------------
DEV void tdm_load_2d(const void* gaddr, unsigned lds_byte_off,
                     unsigned tile_w, unsigned tile_h,
                     unsigned tensor_w, unsigned tensor_h,
                     unsigned stride_elems)
{
  unsigned long long ga = (unsigned long long)(size_t)gaddr;
  u32x4 g0;
  g0[0] = 1u;                                              // count = 1
  g0[1] = lds_byte_off;                                    // lds_addr
  g0[2] = (unsigned)ga;                                    // global_addr[31:0]
  g0[3] = ((unsigned)(ga >> 32) & 0x01FFFFFFu) | (2u << 30); // addr[56:32] | type=2
  i32x8 g1;
  g1[0] = (int)(1u << 16);                                 // data_size = 2 bytes
  g1[1] = (int)((tensor_w & 0xFFFFu) << 16);               // tensor_dim0[15:0]
  g1[2] = (int)((tensor_w >> 16) | ((tensor_h & 0xFFFFu) << 16)); // td0 hi | td1 lo
  g1[3] = (int)((tensor_h >> 16) | (tile_w << 16));        // td1 hi | tile_dim0
  g1[4] = (int)tile_h;                                     // tile_dim1 (tile_dim2=0)
  g1[5] = (int)stride_elems;                               // tensor_dim0_stride[31:0]
  g1[6] = 0;
  g1[7] = 0;
  i32x4 gz4 = {0, 0, 0, 0};
  i32x8 gz8 = {0, 0, 0, 0, 0, 0, 0, 0};
  __builtin_amdgcn_tensor_load_to_lds(g0, g1, gz4, gz4, gz8, 0);
}

// ---------------------------------------------------------------------------
// Generic bf16 WMMA GEMM: C = epi(A[MxK] @ B[KxN]); K%32==0, N%64==0.
// A-tiles: TDM double-buffered (wave 0 issues, TENSORcnt sync; M-tail
// zero-filled by tensor_dim1). B-tiles: coalesced global_load_b128 +
// transposed LDS store so B fragments are two ds_load_b128 per lane.
// Epilogue fuses scale/bias/residual/relu, dual f32+bf16 output.
// Block = 256 thr = 8 waves; tile 64x64x32; wave tile 16x32 (2 WMMAs).
// ---------------------------------------------------------------------------
__global__ void k_gemm(const ush* __restrict__ A, const ush* __restrict__ B,
                       float* __restrict__ Cf, ush* __restrict__ Cb,
                       const float* __restrict__ scale, const float* __restrict__ bias,
                       const float* __restrict__ res,
                       int M, int Nn, int Kd, int relu)
{
  __shared__ __attribute__((aligned(16))) ush As[2][64][32];  // TDM-filled
  __shared__ __attribute__((aligned(16))) ush Bs[64][40];     // Bs[n][k] (transposed)
  const int tid = threadIdx.x;
  const int lane = tid & 31, wid = tid >> 5;
  const int m0 = blockIdx.y * 64, n0 = blockIdx.x * 64;
  const int wm = (wid >> 1) * 16, wn = (wid & 1) * 32;
  floatx8 acc0 = {0.f,0.f,0.f,0.f,0.f,0.f,0.f,0.f};
  floatx8 acc1 = {0.f,0.f,0.f,0.f,0.f,0.f,0.f,0.f};
  const int kb = tid >> 3, nb = (tid & 7) * 8;   // B stage: 32 rows x 64 cols
  const unsigned ldsA0 = (unsigned)(size_t)&As[0][0][0];
  const unsigned ldsA1 = (unsigned)(size_t)&As[1][0][0];
  const unsigned Mrem = (unsigned)(M - m0);

  if (wid == 0)
    tdm_load_2d(A + (size_t)m0 * Kd, ldsA0, 32, 64, (unsigned)Kd, Mrem, (unsigned)Kd);

  int cur = 0;
  for (int kt = 0; kt < Kd; kt += 32) {
    // stage B tile (transposed into LDS)
    ushort8 bv = *reinterpret_cast<const ushort8*>(B + (size_t)(kt + kb)*Nn + n0 + nb);
#pragma unroll
    for (int j = 0; j < 8; ++j) Bs[nb + j][kb] = bv[j];

    const bool more = (kt + 32 < Kd);
    if (more) {
      if (wid == 0)   // TDM prefetch of next A tile into the other buffer
        tdm_load_2d(A + (size_t)m0 * Kd + kt + 32, cur ? ldsA0 : ldsA1,
                    32, 64, (unsigned)(Kd - kt - 32), Mrem, (unsigned)Kd);
      __builtin_prefetch(B + (size_t)(kt + 32 + kb)*Nn + n0 + nb, 0, 1);
      __builtin_amdgcn_s_wait_tensorcnt(1);   // current tile done, next in flight
    } else {
      __builtin_amdgcn_s_wait_tensorcnt(0);
    }
    __syncthreads();

    // A frag (16x32 bf16): lanes 0-15 -> K {0..7,16..23}; lanes 16-31 -> K {8..15,24..31}
    Frag16 fa, fb0, fb1;
    const int arow = wm + (lane & 15);
    const int ak = (lane < 16) ? 0 : 8;
    fa.h[0] = *reinterpret_cast<ushort8*>(&As[cur][arow][ak]);
    fa.h[1] = *reinterpret_cast<ushort8*>(&As[cur][arow][ak + 16]);
    // B frag (32x16 bf16): lanes 0-15 -> K {0..7,16..23}; lanes 16-31 -> K {8..15,24..31}, N = lane&15
    const int bk = (lane < 16) ? 0 : 8;
    const int brow = wn + (lane & 15);
    fb0.h[0] = *reinterpret_cast<ushort8*>(&Bs[brow][bk]);
    fb0.h[1] = *reinterpret_cast<ushort8*>(&Bs[brow][bk + 16]);
    fb1.h[0] = *reinterpret_cast<ushort8*>(&Bs[brow + 16][bk]);
    fb1.h[1] = *reinterpret_cast<ushort8*>(&Bs[brow + 16][bk + 16]);

    acc0 = __builtin_amdgcn_wmma_f32_16x16x32_bf16(false, fa.v, false, fb0.v,
                                                   (short)0, acc0, false, false);
    acc1 = __builtin_amdgcn_wmma_f32_16x16x32_bf16(false, fa.v, false, fb1.v,
                                                   (short)0, acc1, false, false);
    __syncthreads();
    cur ^= 1;
  }

  // C/D layout: VGPR r -> M = r + (lane<16?0:8); N = lane&15
  const int mlo = m0 + wm + ((lane < 16) ? 0 : 8);
  const int nc0 = n0 + wn + (lane & 15);
#pragma unroll
  for (int r = 0; r < 8; ++r) {
    int m = mlo + r;
    if (m >= M) continue;
#pragma unroll
    for (int half = 0; half < 2; ++half) {
      int nc = nc0 + half * 16;
      float val = half ? acc1[r] : acc0[r];
      if (scale) val *= scale[nc];
      if (bias)  val += bias[nc];
      if (res)   val += res[(size_t)m * Nn + nc];
      if (relu)  val = fmaxf(val, 0.f);
      if (Cf) Cf[(size_t)m * Nn + nc] = val;
      if (Cb) Cb[(size_t)m * Nn + nc] = f2bf(val);
    }
  }
}

// ------------------------- conversion / packing ----------------------------
__global__ void k_cvt(ush* __restrict__ dst, const float* __restrict__ src, int count) {
  int i = blockIdx.x * 256 + threadIdx.x;
  if (i < count) dst[i] = f2bf(src[i]);
}
__global__ void k_pack_pad(ush* __restrict__ dst, const float* __restrict__ src,
                           int Kin, int Nc, int count) {
  int i = blockIdx.x * 256 + threadIdx.x;
  if (i >= count) return;
  int k = i / Nc, c = i % Nc;
  dst[i] = (k < Kin) ? f2bf(src[(size_t)k * Nc + c]) : (ush)0;
}
__global__ void k_pack_wT(ush* __restrict__ dst, const float* __restrict__ src,
                          int O, int I) {  // src (O,I) -> dst (I,O)
  int idx = blockIdx.x * 256 + threadIdx.x;
  if (idx >= I * O) return;
  int i = idx / O, o = idx % O;
  dst[idx] = f2bf(src[(size_t)o * I + i]);
}
__global__ void k_pack_conv3(ush* __restrict__ dst, const float* __restrict__ src,
                             int O, int I) {  // src (O,I,3,3,3) -> dst (27*I, O)
  int idx = blockIdx.x * 256 + threadIdx.x;
  if (idx >= 27 * I * O) return;
  int kc = idx / O, o = idx % O;
  int t = kc / I, i = kc % I;
  dst[idx] = f2bf(src[((size_t)o * I + i) * 27 + t]);
}

// ------------------------- edge geometry / features ------------------------
__global__ void k_edge_geom(const float* __restrict__ aff, const int* __restrict__ ei,
                            float* __restrict__ geom) {
  int idx = blockIdx.x * 256 + threadIdx.x;
  if (idx >= NKE) return;
  int n = idx / KK;
  int e = ei[idx];
  float nb0 = aff[e*12+3]  - aff[n*12+3];
  float nb1 = aff[e*12+7]  - aff[n*12+7];
  float nb2 = aff[e*12+11] - aff[n*12+11];
  float dist = sqrtf(nb0*nb0 + nb1*nb1 + nb2*nb2);
  float* g = geom + (size_t)idx * 23;
  g[0] = nb0; g[1] = nb1; g[2] = nb2;
#pragma unroll
  for (int j = 0; j < 10; ++j) {
    float f = expf(-3.9120230054281461f * (float)j * 0.1f); // 50^(-j/10)
    float a = dist * f;
    g[3 + j]  = sinf(a);
    g[13 + j] = cosf(a);
  }
}

__global__ void k_nv_mean(const float* __restrict__ geom, const float* __restrict__ wnv,
                          float* __restrict__ xnef) {
  __shared__ float gs[KK * 23];
  int n = blockIdx.x, t = threadIdx.x;
  for (int i = t; i < KK * 23; i += 64) gs[i] = geom[(size_t)n * KK * 23 + i];
  __syncthreads();
  float acc = 0.f;
  for (int k = 0; k < KK; ++k) {
    float s = 0.f;
#pragma unroll
    for (int j = 0; j < 23; ++j) s += gs[k*23 + j] * wnv[j*64 + t];
    acc += fmaxf(s, 0.f);
  }
  xnef[(size_t)n * 320 + 256 + t] = acc * (1.f / (float)KK);
}

__global__ void k_copy_x(float* __restrict__ xnef, const float* __restrict__ x) {
  int idx = blockIdx.x * 256 + threadIdx.x;
  if (idx >= NN * 256) return;
  int n = idx >> 8, i = idx & 255;
  xnef[(size_t)n * 320 + i] = x[idx];
}

// ------------------------- cryo sampling -----------------------------------
__global__ void k_sample_line(const float* __restrict__ aff, const float* __restrict__ geom,
                              const float* __restrict__ grid, float* __restrict__ rec) {
  int idx = blockIdx.x * 256 + threadIdx.x;
  if (idx >= NKE * 45) return;
  int nk = idx / 45, r = idx % 45;
  int q = r / 9, yy = (r / 3) % 3, xx = r % 3;
  int n = nk / KK;
  const float* g = geom + (size_t)nk * 23;
  float v0 = g[0], v1 = g[1], v2 = g[2];
  float c0 = aff[n*12+3], c1 = aff[n*12+7], c2 = aff[n*12+11];
  float nrm = sqrtf(v0*v0 + v1*v1 + v2*v2) + 1e-8f;
  float u0 = v0/nrm, u1 = v1/nrm, u2 = v2/nrm;
  float a0, a1, a2;
  if (fabsf(u2) < 0.9f) { a0 = 0.f; a1 = 0.f; a2 = 1.f; }
  else                  { a0 = 1.f; a1 = 0.f; a2 = 0.f; }
  float w10 = u1*a2 - u2*a1, w11 = u2*a0 - u0*a2, w12 = u0*a1 - u1*a0;
  float wn = sqrtf(w10*w10 + w11*w11 + w12*w12) + 1e-8f;
  w10 /= wn; w11 /= wn; w12 /= wn;
  float w20 = u1*w12 - u2*w11, w21 = u2*w10 - u0*w12, w22 = u0*w11 - u1*w10;
  float s = -0.5f + 0.25f * (float)q;
  float oy = (float)(yy - 1), ox = (float)(xx - 1);
  float p0 = c0 + v0*s + oy*w10 + ox*w20;
  float p1 = c1 + v1*s + oy*w11 + ox*w21;
  float p2 = c2 + v2*s + oy*w12 + ox*w22;
  rec[idx] = trilin(grid, p0, p1, p2);
}

__global__ void k_convk(const float* __restrict__ rec, const float* __restrict__ ck,
                        float* __restrict__ kfpre) {
  __shared__ float rs[45];
  int nk = blockIdx.x, ce = threadIdx.x;
  if (ce < 45) rs[ce] = rec[(size_t)nk * 45 + ce];
  __syncthreads();
  float w[9];
#pragma unroll
  for (int j = 0; j < 9; ++j) w[j] = ck[ce * 9 + j];
#pragma unroll
  for (int q = 0; q < 5; ++q) {
    float acc = 0.f;
#pragma unroll
    for (int j = 0; j < 9; ++j) acc += rs[q*9 + j] * w[j];
    kfpre[(size_t)nk * CEQ + ce * 5 + q] = acc;  // feature index ce*Q+q
  }
}

__global__ void k_ln_relu_bf16(const float* __restrict__ src, const float* __restrict__ gain,
                               ush* __restrict__ dst, int rowlen) {
  __shared__ float red[256];
  int row = blockIdx.x, t = threadIdx.x;
  const float* sr = src + (size_t)row * rowlen;
  float s = 0.f;
  for (int j = t; j < rowlen; j += 256) s += sr[j];
  red[t] = s; __syncthreads();
  for (int o = 128; o > 0; o >>= 1) { if (t < o) red[t] += red[t+o]; __syncthreads(); }
  float mean = red[0] / (float)rowlen; __syncthreads();
  float s2 = 0.f;
  for (int j = t; j < rowlen; j += 256) { float d = sr[j] - mean; s2 += d*d; }
  red[t] = s2; __syncthreads();
  for (int o = 128; o > 0; o >>= 1) { if (t < o) red[t] += red[t+o]; __syncthreads(); }
  float rstd = rsqrtf(red[0] / (float)rowlen + 1e-5f);
  ush* dr = dst + (size_t)row * rowlen;
  for (int j = t; j < rowlen; j += 256) {
    float v = (sr[j] - mean) * rstd * gain[j];
    dr[j] = f2bf(fmaxf(v, 0.f));
  }
}

// ------------------------- attention ---------------------------------------
__global__ void k_attention(const float* __restrict__ qb, const float* __restrict__ kfp,
                            const float* __restrict__ vb, const int* __restrict__ ei,
                            float* __restrict__ attn, float* __restrict__ newvec) {
  __shared__ float red[64];
  __shared__ float sc[KK];
  __shared__ int es[KK];
  int n = blockIdx.x, a = blockIdx.y, t = threadIdx.x;
  if (t < KK) es[t] = ei[n * KK + t];
  __syncthreads();
  const float* qr = qb + (size_t)n * AHIF + a * IFZ;
  for (int k = 0; k < KK; ++k) {
    const float* kr = kfp + (size_t)(n * KK + k) * AHIF + a * IFZ;
    float p = 0.f;
    for (int i = t; i < IFZ; i += 64) p += qr[i] * kr[i];
    red[t] = p; __syncthreads();
    for (int o = 32; o > 0; o >>= 1) { if (t < o) red[t] += red[t+o]; __syncthreads(); }
    if (t == 0) sc[k] = red[0] * 0.0625f;  // 1/sqrt(256)
    __syncthreads();
  }
  if (t == 0) {
    float mx = -1e30f;
    for (int k = 0; k < KK; ++k) mx = fmaxf(mx, sc[k]);
    float sum = 0.f;
    for (int k = 0; k < KK; ++k) { sc[k] = expf(sc[k] - mx); sum += sc[k]; }
    float inv = 1.f / sum;
    for (int k = 0; k < KK; ++k) {
      sc[k] *= inv;
      attn[(size_t)(n * KK + k) * AHZ + a] = sc[k];
    }
  }
  __syncthreads();
  for (int i = t; i < IFZ; i += 64) {
    float acc = 0.f;
    for (int k = 0; k < KK; ++k)
      acc += sc[k] * vb[(size_t)es[k] * AHIF + a * IFZ + i];
    newvec[(size_t)n * AHIF + a * IFZ + i] = acc;
  }
}

__global__ void k_build_ef(const float* __restrict__ vb, const float* __restrict__ geom,
                           const float* __restrict__ attn, const int* __restrict__ ei,
                           ush* __restrict__ ef) {
  int nk = blockIdx.x, t = threadIdx.x;
  int e = ei[nk];
  ush* er = ef + (size_t)nk * EFW;
  for (int j = t; j < EFW; j += 256) {
    float v;
    if (j < AHIF)           v = vb[(size_t)e * AHIF + j];
    else if (j < AHIF + 23) v = geom[(size_t)nk * 23 + (j - AHIF)];
    else if (j < AHIF + 31) v = attn[(size_t)nk * 8 + (j - AHIF - 23)];
    else                    v = 0.f;   // pad 2079 -> 2080
    er[j] = f2bf(v);
  }
}

__global__ void k_head_out(float* __restrict__ dst, const float* __restrict__ src,
                           const float* __restrict__ wo, const float* __restrict__ bo, int O) {
  __shared__ float red[256];
  int m = blockIdx.x, t = threadIdx.x;
  float v = src[(size_t)m * 256 + t];
  for (int o = 0; o < O; ++o) {
    red[t] = v * wo[t * O + o];
    __syncthreads();
    for (int s = 128; s > 0; s >>= 1) { if (t < s) red[t] += red[t+s]; __syncthreads(); }
    if (t == 0) dst[(size_t)m * O + o] = red[0] + bo[o];
    __syncthreads();
  }
}

// ------------------------- cube path ---------------------------------------
__global__ void k_sample_cube(const float* __restrict__ aff, const float* __restrict__ grid,
                              float* __restrict__ cube) {
  int idx = blockIdx.x * 256 + threadIdx.x;
  if (idx >= NN * P3) return;
  int n = idx / P3, c = idx % P3;
  float oz = (float)(c / 289) - 8.f;
  float oy = (float)((c / 17) % 17) - 8.f;
  float ox = (float)(c % 17) - 8.f;
  const float* A = aff + n * 12;
  float p0 = A[3]  + A[0]*oz + A[1]*oy + A[2]*ox;
  float p1 = A[7]  + A[4]*oz + A[5]*oy + A[6]*ox;
  float p2 = A[11] + A[8]*oz + A[9]*oy + A[10]*ox;
  cube[idx] = trilin(grid, p0, p1, p2);
}

__global__ void k_bn0_c1(const float* __restrict__ cube, const float* __restrict__ w,
                         const float* __restrict__ s, const float* __restrict__ b,
                         ush* __restrict__ dst, int c0) {
  int v = blockIdx.x, o = threadIdx.x;
  float xv = cube[(size_t)c0 * P3 + v];
  float h = fmaxf(xv * w[o] * s[o] + b[o], 0.f);
  dst[(size_t)v * 64 + o] = f2bf(h);
}

__global__ void k_bn0_cd(const float* __restrict__ cube, const float* __restrict__ w,
                         const float* __restrict__ s, const float* __restrict__ b,
                         float* __restrict__ skip, int c0) {
  int r = blockIdx.x, o = threadIdx.x;
  int nl = r / 729, sp = r % 729;
  int oz = sp / 81, oy = (sp / 9) % 9, ox = sp % 9;
  float xv = cube[(size_t)(c0 + nl) * P3 + (2*oz*17 + 2*oy)*17 + 2*ox];
  skip[(size_t)r * 256 + o] = xv * w[o] * s[o] + b[o];
}

__global__ void k_im2col(ush* __restrict__ dst, const ush* __restrict__ src,
                         int Cin, int inD, int outD) {  // 3^3, stride 2, pad 1
  int row = blockIdx.x, t = threadIdx.x;
  int od2 = outD * outD;
  int nl = row / (od2 * outD);
  int r = row % (od2 * outD);
  int oz = r / od2, oy = (r / outD) % outD, ox = r % outD;
  int cols = 27 * Cin;
  ush* dr = dst + (size_t)row * cols;
  for (int j = t; j < cols; j += 256) {
    int tp = j / Cin, ci = j % Cin;
    int dz = tp / 9, dy = (tp / 3) % 3, dx = tp % 3;
    int iz = oz*2 - 1 + dz, iy = oy*2 - 1 + dy, ix = ox*2 - 1 + dx;
    ush v = 0;
    if (iz >= 0 && iz < inD && iy >= 0 && iy < inD && ix >= 0 && ix < inD)
      v = src[((size_t)((nl*inD + iz)*inD + iy)*inD + ix) * Cin + ci];
    dr[j] = v;
  }
}

__global__ void k_gather2(ush* __restrict__ dst, const ush* __restrict__ src,
                          int Cin, int inD, int outD) {  // 1x1x1 stride-2 gather
  int row = blockIdx.x, t = threadIdx.x;
  int od2 = outD * outD;
  int nl = row / (od2 * outD);
  int r = row % (od2 * outD);
  int oz = r / od2, oy = (r / outD) % outD, ox = r % outD;
  size_t base = ((size_t)((nl*inD + 2*oz)*inD + 2*oy)*inD + 2*ox) * Cin;
  for (int j = t; j < Cin; j += 256)
    dst[(size_t)row * Cin + j] = src[base + j];
}

__global__ void k_spatial_mean(float* __restrict__ pooled, const float* __restrict__ src,
                               int c0, int S) {
  int nl = blockIdx.x, c = threadIdx.x;
  float s = 0.f;
  for (int i = 0; i < S; ++i) s += src[(size_t)(nl * S + i) * 256 + c];
  pooled[(size_t)(c0 + nl) * 256 + c] = s / (float)S;
}

// ------------------------- final fuse --------------------------------------
__global__ void k_build_nf(ush* __restrict__ dst, const float* __restrict__ nv,
                           const float* __restrict__ np) {
  int n = blockIdx.x, t = threadIdx.x;
  for (int j = t; j < 4096; j += 256) {
    float v = (j < 2048) ? nv[(size_t)n * 2048 + j] : np[(size_t)n * 2048 + (j - 2048)];
    dst[(size_t)n * 4096 + j] = f2bf(v);
  }
}

__global__ void k_final_ln(float* __restrict__ out, const float* __restrict__ x,
                           const float* __restrict__ nf, const float* __restrict__ g) {
  __shared__ float red[256];
  int n = blockIdx.x, t = threadIdx.x;
  float v = x[(size_t)n*256 + t] + nf[(size_t)n*256 + t] * 0.7071067811865475f;
  red[t] = v; __syncthreads();
  for (int s = 128; s > 0; s >>= 1) { if (t < s) red[t] += red[t+s]; __syncthreads(); }
  float mean = red[0] / 256.f; __syncthreads();
  float d = v - mean; red[t] = d * d; __syncthreads();
  for (int s = 128; s > 0; s >>= 1) { if (t < s) red[t] += red[t+s]; __syncthreads(); }
  float rstd = rsqrtf(red[0] / 256.f + 1e-5f);
  out[(size_t)n*256 + t] = (v - mean) * rstd * g[t];
}

__global__ void k_write_ei(float* __restrict__ dst, const int* __restrict__ ei) {
  int i = blockIdx.x * 256 + threadIdx.x;
  if (i < NKE) dst[i] = (float)ei[i];
}

// ===========================================================================
extern "C" void kernel_launch(void* const* d_in, const int* in_sizes, int n_in,
                              void* d_out, int out_size, void* d_ws, size_t ws_size,
                              hipStream_t stream)
{
  (void)in_sizes; (void)n_in; (void)out_size; (void)ws_size;
  const float* x    = (const float*)d_in[0];
  const float* aff  = (const float*)d_in[1];
  const float* grid = (const float*)d_in[3];
  const int*   ei   = (const int*)d_in[4];
  auto P = [&](int i) { return (const float*)d_in[i]; };
  // params flattened in _make_params insertion order, starting at d_in[5]
  const int I_WNV=5, I_WQ=6, I_WV=7, I_CONVK=8, I_GLNK=9, I_WK=10;
  const int I_BN0=11;             // 12 entries per block: c1,s1,b1,c2,s2,b2,c3,s3,b3,cd,sd,bd
  const int I_WPV=59;
  const int I_EH0=60, I_EH1=65, I_EH2=69;   // eh0: w1,b1,w2,b2,wp
  const int I_AH0=73, I_AH1=78, I_AH2=82;
  const int I_EHWO=86, I_EHBO=87, I_AHWO=88, I_AHBO=89, I_WAG=90, I_GEN=91;

  float* dout = (float*)d_out;
  const int OUT_EI = 131072, OUT_EL = 141312, OUT_AA = 151552;

  char* base = (char*)d_ws; size_t off = 0;
  auto alloc = [&](size_t bytes) -> void* {
    off = (off + 255) & ~(size_t)255;
    void* p = base + off; off += bytes; return p;
  };

  float* geom   = (float*)alloc((size_t)NKE*23*4);
  float* xnef   = (float*)alloc((size_t)NN*320*4);
  ush*   xneb   = (ush*)  alloc((size_t)NN*320*2);
  float* rec    = (float*)alloc((size_t)NKE*45*4);
  float* kfpre  = (float*)alloc((size_t)NKE*CEQ*4);
  ush*   kfact  = (ush*)  alloc((size_t)NKE*CEQ*2);
  float* kfp    = (float*)alloc((size_t)NKE*AHIF*4);
  float* qb     = (float*)alloc((size_t)NN*AHIF*4);
  float* vb     = (float*)alloc((size_t)NN*AHIF*4);
  float* attn   = (float*)alloc((size_t)NKE*8*4);
  float* newvec = (float*)alloc((size_t)NN*AHIF*4);
  ush*   efb    = (ush*)  alloc((size_t)NKE*EFW*2);
  ush*   t1b    = (ush*)  alloc((size_t)NKE*256*2);
  float* sfb    = (float*)alloc((size_t)NKE*256*4);
  float* hAf    = (float*)alloc((size_t)NKE*256*4);
  ush*   hAb    = (ush*)  alloc((size_t)NKE*256*2);
  float* hBf    = (float*)alloc((size_t)NKE*256*4);
  ush*   hBb    = (ush*)  alloc((size_t)NKE*256*2);
  float* cube   = (float*)alloc((size_t)NN*P3*4);
  ush*   h1b    = (ush*)  alloc((size_t)32*P3*64*2);
  ush*   im2    = (ush*)  alloc((size_t)32*729*1728*2);
  ush*   c2ob   = (ush*)  alloc((size_t)32*729*64*2);
  float* skipf  = (float*)alloc((size_t)32*729*256*4);
  ush*   gatb   = (ush*)  alloc((size_t)32*125*256*2);
  float* bAf    = (float*)alloc((size_t)32*729*256*4);
  ush*   bAb    = (ush*)  alloc((size_t)32*729*256*2);
  float* bBf    = (float*)alloc((size_t)32*729*256*4);
  ush*   bBb    = (ush*)  alloc((size_t)32*729*256*2);
  float* pooled = (float*)alloc((size_t)NN*256*4);
  ush*   poolb  = (ush*)  alloc((size_t)NN*256*2);
  float* nptf   = (float*)alloc((size_t)NN*AHIF*4);
  ush*   nptb   = (ush*)  alloc((size_t)NN*AHIF*2);
  ush*   nfb    = (ush*)  alloc((size_t)NN*4096*2);
  float* nfo    = (float*)alloc((size_t)NN*256*4);
  // bf16 weights
  ush* wqb    = (ush*)alloc((size_t)320*2048*2);
  ush* wvb    = (ush*)alloc((size_t)320*2048*2);
  ush* wkb    = (ush*)alloc((size_t)1280*2048*2);
  ush* eh0w1b = (ush*)alloc((size_t)EFW*256*2);
  ush* eh0wpb = (ush*)alloc((size_t)EFW*256*2);
  ush* eh0w2b = (ush*)alloc((size_t)256*256*2);
  ush* eh1w1b = (ush*)alloc((size_t)256*256*2);
  ush* eh1w2b = (ush*)alloc((size_t)256*256*2);
  ush* eh2w1b = (ush*)alloc((size_t)256*256*2);
  ush* eh2w2b = (ush*)alloc((size_t)256*256*2);
  ush* ah0w1b = (ush*)alloc((size_t)2048*256*2);
  ush* ah0wpb = (ush*)alloc((size_t)2048*256*2);
  ush* ah0w2b = (ush*)alloc((size_t)256*256*2);
  ush* ah1w1b = (ush*)alloc((size_t)256*256*2);
  ush* ah1w2b = (ush*)alloc((size_t)256*256*2);
  ush* ah2w1b = (ush*)alloc((size_t)256*256*2);
  ush* ah2w2b = (ush*)alloc((size_t)256*256*2);
  ush* wpvb   = (ush*)alloc((size_t)256*2048*2);
  ush* wagb   = (ush*)alloc((size_t)4096*256*2);
  ush* c2wb[4]; ush* c3wb[4]; ush* c1wb[4]; ush* cdwb[4];
  for (int i = 0; i < 4; ++i) {
    c2wb[i] = (ush*)alloc((size_t)1728*64*2);
    c3wb[i] = (ush*)alloc((size_t)64*256*2);
    c1wb[i] = (i >= 1) ? (ush*)alloc((size_t)256*64*2) : nullptr;
    cdwb[i] = (i >= 1) ? (ush*)alloc((size_t)256*256*2) : nullptr;
  }

  auto CVT = [&](ush* dst, const float* src, size_t cnt) {
    k_cvt<<<(unsigned)((cnt + 255) / 256), 256, 0, stream>>>(dst, src, (int)cnt);
  };
  auto GEMM = [&](const ush* A, const ush* B, float* Cf, ush* Cb,
                  const float* sc, const float* bia, const float* rs,
                  int M, int Nc, int Kd, int relu) {
    dim3 g((unsigned)(Nc / 64), (unsigned)((M + 63) / 64));
    k_gemm<<<g, 256, 0, stream>>>(A, B, Cf, Cb, sc, bia, rs, M, Nc, Kd, relu);
  };

  // ---------------- weight packing (bf16) ----------------
  CVT(wqb, P(I_WQ), (size_t)320*2048);
  CVT(wvb, P(I_WV), (size_t)320*2048);
  CVT(wkb, P(I_WK), (size_t)1280*2048);
  k_pack_pad<<<(EFW*256+255)/256, 256, 0, stream>>>(eh0w1b, P(I_EH0+0), 2079, 256, EFW*256);
  k_pack_pad<<<(EFW*256+255)/256, 256, 0, stream>>>(eh0wpb, P(I_EH0+4), 2079, 256, EFW*256);
  CVT(eh0w2b, P(I_EH0+2), 65536);
  CVT(eh1w1b, P(I_EH1+0), 65536); CVT(eh1w2b, P(I_EH1+2), 65536);
  CVT(eh2w1b, P(I_EH2+0), 65536); CVT(eh2w2b, P(I_EH2+2), 65536);
  CVT(ah0w1b, P(I_AH0+0), (size_t)2048*256);
  CVT(ah0wpb, P(I_AH0+4), (size_t)2048*256);
  CVT(ah0w2b, P(I_AH0+2), 65536);
  CVT(ah1w1b, P(I_AH1+0), 65536); CVT(ah1w2b, P(I_AH1+2), 65536);
  CVT(ah2w1b, P(I_AH2+0), 65536); CVT(ah2w2b, P(I_AH2+2), 65536);
  CVT(wpvb, P(I_WPV), (size_t)256*2048);
  CVT(wagb, P(I_WAG), (size_t)4096*256);
  for (int i = 0; i < 4; ++i) {
    int Ib = I_BN0 + i * 12;
    k_pack_conv3<<<(27*64*64+255)/256, 256, 0, stream>>>(c2wb[i], P(Ib+3), 64, 64);
    k_pack_wT<<<(64*256+255)/256, 256, 0, stream>>>(c3wb[i], P(Ib+6), 256, 64);
    if (i >= 1) {
      k_pack_wT<<<(256*64+255)/256, 256, 0, stream>>>(c1wb[i], P(Ib+0), 64, 256);
      k_pack_wT<<<(256*256+255)/256, 256, 0, stream>>>(cdwb[i], P(Ib+9), 256, 256);
    }
  }

  // ---------------- node/edge features ----------------
  k_edge_geom<<<(NKE+255)/256, 256, 0, stream>>>(aff, ei, geom);
  k_copy_x<<<(NN*256+255)/256, 256, 0, stream>>>(xnef, x);
  k_nv_mean<<<NN, 64, 0, stream>>>(geom, P(I_WNV), xnef);
  CVT(xneb, xnef, (size_t)NN*320);
  k_sample_line<<<(NKE*45+255)/256, 256, 0, stream>>>(aff, geom, grid, rec);
  k_convk<<<NKE, 256, 0, stream>>>(rec, P(I_CONVK), kfpre);
  k_ln_relu_bf16<<<NKE, 256, 0, stream>>>(kfpre, P(I_GLNK), kfact, CEQ);

  GEMM(xneb, wqb, qb, nullptr, nullptr, nullptr, nullptr, NN, AHIF, 320, 0);
  GEMM(xneb, wvb, vb, nullptr, nullptr, nullptr, nullptr, NN, AHIF, 320, 0);
  GEMM(kfact, wkb, kfp, nullptr, nullptr, nullptr, nullptr, NKE, AHIF, CEQ, 0);

  { dim3 ga(NN, AHZ); k_attention<<<ga, 64, 0, stream>>>(qb, kfp, vb, ei, attn, newvec); }
  k_build_ef<<<NKE, 256, 0, stream>>>(vb, geom, attn, ei, efb);

  // ---------------- edge head (3x fcres + linear) ----------------
  GEMM(efb, eh0w1b, nullptr, t1b, nullptr, P(I_EH0+1), nullptr, NKE, 256, EFW, 1);
  GEMM(efb, eh0wpb, sfb, nullptr, nullptr, nullptr, nullptr, NKE, 256, EFW, 0);
  GEMM(t1b, eh0w2b, hAf, hAb, nullptr, P(I_EH0+3), sfb, NKE, 256, 256, 1);
  GEMM(hAb, eh1w1b, nullptr, t1b, nullptr, P(I_EH1+1), nullptr, NKE, 256, 256, 1);
  GEMM(t1b, eh1w2b, hBf, hBb, nullptr, P(I_EH1+3), hAf, NKE, 256, 256, 1);
  GEMM(hBb, eh2w1b, nullptr, t1b, nullptr, P(I_EH2+1), nullptr, NKE, 256, 256, 1);
  GEMM(t1b, eh2w2b, hAf, nullptr, nullptr, P(I_EH2+3), hBf, NKE, 256, 256, 1);
  k_head_out<<<NKE, 256, 0, stream>>>(dout + OUT_EL, hAf, P(I_EHWO), P(I_EHBO), 1);

  // ---------------- cube + bottleneck CNN (channels-last, chunked) --------
  k_sample_cube<<<(NN*P3+255)/256, 256, 0, stream>>>(aff, grid, cube);
  for (int c0 = 0; c0 < NN; c0 += 32) {
    k_bn0_c1<<<32*P3, 64, 0, stream>>>(cube, P(I_BN0+0), P(I_BN0+1), P(I_BN0+2), h1b, c0);
    k_bn0_cd<<<32*729, 256, 0, stream>>>(cube, P(I_BN0+9), P(I_BN0+10), P(I_BN0+11), skipf, c0);
    k_im2col<<<32*729, 256, 0, stream>>>(im2, h1b, 64, 17, 9);
    GEMM(im2, c2wb[0], nullptr, c2ob, P(I_BN0+4), P(I_BN0+5), nullptr, 32*729, 64, 1728, 1);
    GEMM(c2ob, c3wb[0], bAf, bAb, P(I_BN0+7), P(I_BN0+8), skipf, 32*729, 256, 64, 1);

    float* inf = bAf; ush* inb = bAb; float* outf = bBf; ush* outb = bBb;
    int inD = 9;
    for (int bi = 1; bi < 4; ++bi) {
      int outD = inD / 2 + 1;                         // 9->5->3->2
      int rin = 32 * inD * inD * inD;
      int rout = 32 * outD * outD * outD;
      int Ib = I_BN0 + bi * 12;
      GEMM(inb, c1wb[bi], nullptr, h1b, P(Ib+1), P(Ib+2), nullptr, rin, 64, 256, 1);
      k_im2col<<<(unsigned)rout, 256, 0, stream>>>(im2, h1b, 64, inD, outD);
      GEMM(im2, c2wb[bi], nullptr, c2ob, P(Ib+4), P(Ib+5), nullptr, rout, 64, 1728, 1);
      k_gather2<<<(unsigned)rout, 256, 0, stream>>>(gatb, inb, 256, inD, outD);
      GEMM(gatb, cdwb[bi], skipf, nullptr, P(Ib+10), P(Ib+11), nullptr, rout, 256, 256, 0);
      GEMM(c2ob, c3wb[bi], outf, outb, P(Ib+7), P(Ib+8), skipf, rout, 256, 64, 1);
      { float* tf = inf; inf = outf; outf = tf; ush* tb = inb; inb = outb; outb = tb; }
      inD = outD;
    }
    k_spatial_mean<<<32, 256, 0, stream>>>(pooled, inf, c0, 8);
  }

  CVT(poolb, pooled, (size_t)NN*256);
  GEMM(poolb, wpvb, nptf, nptb, nullptr, nullptr, nullptr, NN, AHIF, 256, 0);

  // ---------------- aa head ----------------
  GEMM(nptb, ah0w1b, nullptr, t1b, nullptr, P(I_AH0+1), nullptr, NN, 256, 2048, 1);
  GEMM(nptb, ah0wpb, sfb, nullptr, nullptr, nullptr, nullptr, NN, 256, 2048, 0);
  GEMM(t1b, ah0w2b, hAf, hAb, nullptr, P(I_AH0+3), sfb, NN, 256, 256, 1);
  GEMM(hAb, ah1w1b, nullptr, t1b, nullptr, P(I_AH1+1), nullptr, NN, 256, 256, 1);
  GEMM(t1b, ah1w2b, hBf, hBb, nullptr, P(I_AH1+3), hAf, NN, 256, 256, 1);
  GEMM(hBb, ah2w1b, nullptr, t1b, nullptr, P(I_AH2+1), nullptr, NN, 256, 256, 1);
  GEMM(t1b, ah2w2b, hAf, nullptr, nullptr, P(I_AH2+3), hBf, NN, 256, 256, 1);
  k_head_out<<<NN, 256, 0, stream>>>(dout + OUT_AA, hAf, P(I_AHWO), P(I_AHBO), 20);

  // ---------------- final aggregate + LN ----------------
  k_build_nf<<<NN, 256, 0, stream>>>(nfb, newvec, nptf);
  GEMM(nfb, wagb, nfo, nullptr, nullptr, nullptr, nullptr, NN, 256, 4096, 0);
  k_final_ln<<<NN, 256, 0, stream>>>(dout, x, nfo, P(I_GEN));
  k_write_ei<<<(NKE+255)/256, 256, 0, stream>>>(dout + OUT_EI, ei);
}
